// GAT_Embedding_5540507812193
// MI455X (gfx1250) — compile-verified, matrix operand order checked by reading, not run on hardware
//
#include <hip/hip_runtime.h>
#include <hip/hip_bf16.h>

#define GAT_N    50000
#define GAT_E    800000
#define GAT_K    128          // input feature dim of both GEMMs
#define GAT_HID  16
#define GAT_H    8

typedef __attribute__((ext_vector_type(16))) _Float16 v16h;
typedef __attribute__((ext_vector_type(8)))  float    v8f;

static __host__ __device__ inline int cdiv(long long a, long long b) { return (int)((a + b - 1) / b); }

// ---------------------------------------------------------------------------
// order-preserving float <-> uint map for atomicMax-based segment max
// ---------------------------------------------------------------------------
__device__ inline unsigned f2ord(float f) {
    unsigned u = __float_as_uint(f);
    return (u & 0x80000000u) ? ~u : (u | 0x80000000u);
}
__device__ inline float ord2f(unsigned u) {
    unsigned v = (u & 0x80000000u) ? (u & 0x7FFFFFFFu) : ~u;
    return __uint_as_float(v);
}

// ---------------------------------------------------------------------------
// WMMA GEMM: C[M,:] = A[M,128] @ B[Nout,128]^T   (fp32 in, f16 WMMA, fp32 out)
// One wave (32 threads) per 16x16 output tile; K=128 fully unrolled -> 4 WMMAs.
// A/B operand layout per CDNA5 ISA 7.12.2 (16-bit A 16x32):
//   lane<16:  row=lane,    halves 0..7 = K(+0..7),  halves 8..15 = K(+16..23)
//   lane>=16: row=lane-16, halves 0..7 = K(+8..15), halves 8..15 = K(+24..31)
// D layout: lane<16 -> rows 0..7 (VGPR i = row i), lane>=16 -> rows 8..15.
// ---------------------------------------------------------------------------
__global__ void wmma_gemm_abT(const float* __restrict__ A,
                              const float* __restrict__ B,
                              float* __restrict__ C,
                              int ldc) {
    const int lane  = threadIdx.x;            // 0..31
    const int mBase = blockIdx.x * 16;
    const int nBase = blockIdx.y * 16;
    const int rsel  = lane & 15;
    const int kOff  = (lane < 16) ? 0 : 8;

    const float* ap = A + (size_t)(mBase + rsel) * GAT_K + kOff;
    const float* bp = B + (size_t)(nBase + rsel) * GAT_K + kOff;

    v8f acc = {};
#pragma unroll
    for (int kk = 0; kk < GAT_K; kk += 32) {
        v16h a, b;
#pragma unroll
        for (int i = 0; i < 8; ++i) {
            a[i]     = (_Float16)ap[kk + i];
            a[i + 8] = (_Float16)ap[kk + i + 16];
            b[i]     = (_Float16)bp[kk + i];
            b[i + 8] = (_Float16)bp[kk + i + 16];
        }
        acc = __builtin_amdgcn_wmma_f32_16x16x32_f16(
            /*neg_a=*/false, a, /*neg_b=*/false, b,
            /*c_mod=*/(short)0, acc, /*reuse_a=*/false, /*reuse_b=*/false);
    }

    const int col     = nBase + rsel;
    const int rowBase = mBase + ((lane < 16) ? 0 : 8);
#pragma unroll
    for (int i = 0; i < 8; ++i)
        C[(size_t)(rowBase + i) * ldc + col] = acc[i];
}

// ---------------------------------------------------------------------------
// per-node attention scores: a_src[n,h] = sum_c h[n,h,c]*att_src[h,c] (same for dst)
// ---------------------------------------------------------------------------
__global__ void attn_scores(const float* __restrict__ h,
                            const float* __restrict__ att_s,
                            const float* __restrict__ att_d,
                            float* __restrict__ as_out,
                            float* __restrict__ ad_out,
                            int n, int heads, int c) {
    int idx = blockIdx.x * blockDim.x + threadIdx.x;
    if (idx >= n * heads) return;
    int hh = idx % heads;
    const float* hp = h + (size_t)idx * c;     // h is [n, heads*c] contiguous
    float s = 0.f, d = 0.f;
    for (int i = 0; i < c; ++i) {
        float v = hp[i];
        s += v * att_s[hh * c + i];
        d += v * att_d[hh * c + i];
    }
    as_out[idx] = s;
    ad_out[idx] = d;
}

// ---------------------------------------------------------------------------
// edge pass 1: alpha = leaky_relu(a_src[src]+a_dst[dst], 0.2); segment atomic max
// ---------------------------------------------------------------------------
__global__ void edge_alpha_max(const int* __restrict__ src, const int* __restrict__ dst,
                               const float* __restrict__ as, const float* __restrict__ ad,
                               float* __restrict__ alpha, unsigned* __restrict__ amax,
                               int E, int heads) {
    int idx = blockIdx.x * blockDim.x + threadIdx.x;
    if (idx >= E * heads) return;
    int e = idx / heads, hh = idx - e * heads;
    int s = src[e], d = dst[e];
    float a = as[s * heads + hh] + ad[d * heads + hh];
    a = (a > 0.f) ? a : 0.2f * a;
    alpha[idx] = a;
    atomicMax(&amax[d * heads + hh], f2ord(a));
}

// ---------------------------------------------------------------------------
// edge pass 2: ex = exp(alpha - amax[dst]); segment atomic sum (in-place over alpha)
// ---------------------------------------------------------------------------
__global__ void edge_exp_sum(const int* __restrict__ dst,
                             float* __restrict__ alpha,
                             const unsigned* __restrict__ amax,
                             float* __restrict__ denom,
                             int E, int heads) {
    int idx = blockIdx.x * blockDim.x + threadIdx.x;
    if (idx >= E * heads) return;
    int e = idx / heads, hh = idx - e * heads;
    int d = dst[e];
    float m = ord2f(amax[d * heads + hh]);
    if (!isfinite(m)) m = 0.f;            // matches reference's where(isfinite, amax, 0)
    float ex = expf(alpha[idx] - m);
    alpha[idx] = ex;
    atomicAdd(&denom[d * heads + hh], ex);
}

// ---------------------------------------------------------------------------
// edge pass 2.5: attn = ex / (denom[dst,h] + 1e-16)   (in-place over ex)
// ---------------------------------------------------------------------------
__global__ void edge_normalize(const int* __restrict__ dst,
                               float* __restrict__ ex,
                               const float* __restrict__ denom,
                               int E, int heads) {
    int idx = blockIdx.x * blockDim.x + threadIdx.x;
    if (idx >= E * heads) return;
    int e = idx / heads, hh = idx - e * heads;
    ex[idx] = ex[idx] / (denom[dst[e] * heads + hh] + 1e-16f);
}

// ---------------------------------------------------------------------------
// edge pass 3 (float4): agg[dst, j..j+3] += h[src, j..j+3] * attn[e,h]
// one thread per (edge, 4 contiguous features); c multiple of 4 so a float4
// never straddles two heads. Scatter target (N*F*4 B <= 25.6 MB) is L2-resident.
// ---------------------------------------------------------------------------
__global__ void edge_aggregate_v4(const int* __restrict__ src, const int* __restrict__ dst,
                                  const float* __restrict__ h,
                                  const float* __restrict__ attn,
                                  float* __restrict__ agg,
                                  int E, int heads, int c) {
    const int F = heads * c;
    const int Q = F >> 2;                       // float4 chunks per edge
    long long idx = (long long)blockIdx.x * blockDim.x + threadIdx.x;
    if (idx >= (long long)E * Q) return;
    int e = (int)(idx / Q);
    int q = (int)(idx - (long long)e * Q);
    int j = q << 2;
    int hh = j / c;
    int s = src[e], d = dst[e];
    float a = attn[(size_t)e * heads + hh];
    const float4 hv = *(const float4*)(h + (size_t)s * F + j);
    float* ap = agg + (size_t)d * F + j;
    atomicAdd(ap + 0, hv.x * a);
    atomicAdd(ap + 1, hv.y * a);
    atomicAdd(ap + 2, hv.z * a);
    atomicAdd(ap + 3, hv.w * a);
}

// ---------------------------------------------------------------------------
// epilogue: out = elu( BN(agg + bias) + skip )     (may be in-place: out==agg)
// ---------------------------------------------------------------------------
__global__ void epilogue(const float* __restrict__ agg,
                         const float* __restrict__ bias,
                         const float* __restrict__ g, const float* __restrict__ beta,
                         const float* __restrict__ mean, const float* __restrict__ var,
                         const float* __restrict__ skip,
                         float* __restrict__ out,
                         int n, int F) {
    int idx = blockIdx.x * blockDim.x + threadIdx.x;
    if (idx >= n * F) return;
    int j = idx % F;
    float v = agg[idx] + bias[j];
    v = (v - mean[j]) * rsqrtf(var[j] + 1e-5f) * g[j] + beta[j];
    v += skip[idx];
    out[idx] = (v > 0.f) ? v : (expf(v) - 1.f);
}

// ---------------------------------------------------------------------------
extern "C" void kernel_launch(void* const* d_in, const int* in_sizes, int n_in,
                              void* d_out, int out_size, void* d_ws, size_t ws_size,
                              hipStream_t stream) {
    (void)in_sizes; (void)n_in; (void)out_size; (void)ws_size;

    const float* x      = (const float*)d_in[0];
    const int*   ei     = (const int*)  d_in[1];
    const float* w0     = (const float*)d_in[2];
    const float* asrc0  = (const float*)d_in[3];
    const float* adst0  = (const float*)d_in[4];
    const float* b0     = (const float*)d_in[5];
    const float* skip0  = (const float*)d_in[6];
    const float* bn0_g  = (const float*)d_in[7];
    const float* bn0_b  = (const float*)d_in[8];
    const float* bn0_m  = (const float*)d_in[9];
    const float* bn0_v  = (const float*)d_in[10];
    const float* w1     = (const float*)d_in[11];
    const float* asrc1  = (const float*)d_in[12];
    const float* adst1  = (const float*)d_in[13];
    const float* b1     = (const float*)d_in[14];
    const float* skip1  = (const float*)d_in[15];
    const float* bn1_g  = (const float*)d_in[16];
    const float* bn1_b  = (const float*)d_in[17];
    const float* bn1_m  = (const float*)d_in[18];
    const float* bn1_v  = (const float*)d_in[19];

    const int* src = ei;            // edge_index[0]
    const int* dst = ei + GAT_E;    // edge_index[1]

    const int N = GAT_N, E = GAT_E;
    const int F0 = GAT_H * GAT_HID;                // 128

    // ---- workspace carving (floats) ----
    float* ws = (float*)d_ws;
    size_t o = 0;
    float*    H0    = ws + o; o += (size_t)N * F0;    // x @ w0^T
    float*    SK0   = ws + o; o += (size_t)N * F0;    // x @ skip0^T
    float*    AS0   = ws + o; o += (size_t)N * GAT_H;
    float*    AD0   = ws + o; o += (size_t)N * GAT_H;
    unsigned* AMAX0 = (unsigned*)(ws + o); o += (size_t)N * GAT_H;
    float*    DEN0  = ws + o; o += (size_t)N * GAT_H;
    float*    EX0   = ws + o; o += (size_t)E * GAT_H; // alpha -> ex -> attn (in-place)
    float*    AGG0  = ws + o; o += (size_t)N * F0;    // agg -> h_l0 (in-place)
    // layer-1 buffers reuse the (then-dead) H0 region
    size_t o1 = (size_t)(H0 - ws);
    float*    H1    = ws + o1; o1 += (size_t)N * GAT_HID;
    float*    SK1   = ws + o1; o1 += (size_t)N * GAT_HID;
    float*    AS1   = ws + o1; o1 += (size_t)N;
    float*    AD1   = ws + o1; o1 += (size_t)N;
    unsigned* AMAX1 = (unsigned*)(ws + o1); o1 += (size_t)N;
    float*    DEN1  = ws + o1; o1 += (size_t)N;
    float*    EX1   = ws + o1; o1 += (size_t)E;
    float*    AGG1  = ws + o1; o1 += (size_t)N * GAT_HID;

    const int TB = 256;

    // ================= layer 0 =================
    hipMemsetAsync(AMAX0, 0, (size_t)N * GAT_H * 4, stream);
    hipMemsetAsync(DEN0,  0, (size_t)N * GAT_H * 4, stream);
    hipMemsetAsync(AGG0,  0, (size_t)N * F0 * 4,    stream);

    {   // H0 = x @ w0^T ; SK0 = x @ skip0^T   (M=50000 -> 3125 tiles, Nout=128 -> 8 tiles)
        dim3 grid(N / 16, F0 / 16);
        wmma_gemm_abT<<<grid, 32, 0, stream>>>(x, w0,    H0,  F0);
        wmma_gemm_abT<<<grid, 32, 0, stream>>>(x, skip0, SK0, F0);
    }
    attn_scores<<<cdiv((long long)N * GAT_H, TB), TB, 0, stream>>>(
        H0, asrc0, adst0, AS0, AD0, N, GAT_H, GAT_HID);
    edge_alpha_max<<<cdiv((long long)E * GAT_H, TB), TB, 0, stream>>>(
        src, dst, AS0, AD0, EX0, AMAX0, E, GAT_H);
    edge_exp_sum<<<cdiv((long long)E * GAT_H, TB), TB, 0, stream>>>(
        dst, EX0, AMAX0, DEN0, E, GAT_H);
    edge_normalize<<<cdiv((long long)E * GAT_H, TB), TB, 0, stream>>>(
        dst, EX0, DEN0, E, GAT_H);
    edge_aggregate_v4<<<cdiv((long long)E * (F0 / 4), TB), TB, 0, stream>>>(
        src, dst, H0, EX0, AGG0, E, GAT_H, GAT_HID);
    // AGG0 becomes h_l0 in-place
    epilogue<<<cdiv((long long)N * F0, TB), TB, 0, stream>>>(
        AGG0, b0, bn0_g, bn0_b, bn0_m, bn0_v, SK0, AGG0, N, F0);

    // ================= layer 1 =================
    hipMemsetAsync(AMAX1, 0, (size_t)N * 4, stream);
    hipMemsetAsync(DEN1,  0, (size_t)N * 4, stream);
    hipMemsetAsync(AGG1,  0, (size_t)N * GAT_HID * 4, stream);

    {   // H1 = h_l0 @ w1^T ; SK1 = h_l0 @ skip1^T   (Nout=16 -> 1 tile)
        dim3 grid(N / 16, GAT_HID / 16);
        wmma_gemm_abT<<<grid, 32, 0, stream>>>(AGG0, w1,    H1,  GAT_HID);
        wmma_gemm_abT<<<grid, 32, 0, stream>>>(AGG0, skip1, SK1, GAT_HID);
    }
    attn_scores<<<cdiv((long long)N, TB), TB, 0, stream>>>(
        H1, asrc1, adst1, AS1, AD1, N, 1, GAT_HID);
    edge_alpha_max<<<cdiv((long long)E, TB), TB, 0, stream>>>(
        src, dst, AS1, AD1, EX1, AMAX1, E, 1);
    edge_exp_sum<<<cdiv((long long)E, TB), TB, 0, stream>>>(
        dst, EX1, AMAX1, DEN1, E, 1);
    edge_normalize<<<cdiv((long long)E, TB), TB, 0, stream>>>(
        dst, EX1, DEN1, E, 1);
    edge_aggregate_v4<<<cdiv((long long)E * (GAT_HID / 4), TB), TB, 0, stream>>>(
        src, dst, H1, EX1, AGG1, E, 1, GAT_HID);
    epilogue<<<cdiv((long long)N * GAT_HID, TB), TB, 0, stream>>>(
        AGG1, b1, bn1_g, bn1_b, bn1_m, bn1_v, SK1, (float*)d_out, N, GAT_HID);
}
